// Attention_64699387347309
// MI455X (gfx1250) — compile-verified, matrix operand order checked by reading, not run on hardware
//
#include <hip/hip_runtime.h>
#include <hip/hip_bf16.h>

#define S_TOK 2048
#define HEADS 32
#define HDIM 128
#define HID 4096
#define NVIS 576
#define QKSCALE 0.08838834764831845f  // 1/sqrt(128)

typedef __attribute__((ext_vector_type(16))) __bf16 v16bf;
typedef __attribute__((ext_vector_type(8)))  __bf16 v8bf;
typedef __attribute__((ext_vector_type(8)))  float  v8f;

__device__ __forceinline__ unsigned short f2bf(float x) {
    unsigned u = __float_as_uint(x);
    u += 0x7FFFu + ((u >> 16) & 1u);          // round-to-nearest-even
    return (unsigned short)(u >> 16);
}

__device__ __forceinline__ v16bf mk16(v8bf lo, v8bf hi) {
    return __builtin_shufflevector(lo, hi, 0,1,2,3,4,5,6,7,8,9,10,11,12,13,14,15);
}

// CDNA5 async DMA: 16B global -> LDS per lane, tracked by ASYNCcnt
__device__ __forceinline__ void async_load_b128(unsigned lds_off, const void* gaddr) {
    asm volatile("global_load_async_to_lds_b128 %0, %1, off"
                 :: "v"(lds_off), "v"((unsigned long long)(uintptr_t)gaddr)
                 : "memory");
}

// ---------------- prepass: f32 -> bf16 ----------------
__global__ void cvt_bf16(const float* __restrict__ x, unsigned short* __restrict__ y, int n) {
    int i = blockIdx.x * blockDim.x + threadIdx.x;
    if (i < n) y[i] = f2bf(x[i]);
}

// W[K,N] f32 -> Wt[N,K] bf16 (LDS tile transpose, coalesced both sides)
__global__ void transpose_bf16(const float* __restrict__ W, unsigned short* __restrict__ Wt,
                               int K, int N) {
    __shared__ unsigned short tile[32][33];
    int nb = blockIdx.x * 32, kb = blockIdx.y * 32;
    int tx = threadIdx.x & 31, ty = threadIdx.x >> 5;   // blockDim = 256
    for (int i = ty; i < 32; i += 8)
        tile[i][tx] = f2bf(W[(size_t)(kb + i) * N + nb + tx]);
    __syncthreads();
    for (int i = ty; i < 32; i += 8)
        Wt[(size_t)(nb + i) * K + kb + tx] = tile[tx][i];
}

// ---------------- routed GEMM: C[M,N] = A[M,K](bf16) x Wt[N,K]^T (bf16) ----------------
// 64x64 block tile, double-buffered async global->LDS staging, 4 waves,
// block-level expert select (NVIS = 9*64 so 64-row blocks are modality-uniform)
__global__ void gemm_routed(const unsigned short* __restrict__ A,
                            const unsigned short* __restrict__ Wt_lang,
                            const unsigned short* __restrict__ Wt_vis,
                            const float* __restrict__ bias_vis,   // [N] or nullptr
                            float* __restrict__ C, int N, int K) {
    __shared__ alignas(16) unsigned short lds_a[2][64 * 32];   // 4KB x2
    __shared__ alignas(16) unsigned short lds_b[2][64 * 32];   // 4KB x2

    const int tid   = threadIdx.x;        // 0..127
    const int lane  = tid & 31;
    const int wave  = tid >> 5;
    const int row0  = blockIdx.x * 64;
    const int col0  = blockIdx.y * 64;
    const bool vis  = row0 < NVIS;
    const unsigned short* W = vis ? Wt_vis : Wt_lang;

    const int r     = lane & 15;          // fragment row/col within 16-tile
    const int koff  = (lane >> 4) << 3;   // 0 or 8: per-half K offset
    const int mhalf = (lane >> 4) << 3;   // C-layout row offset

    v8f acc[4] = {};

    // stage one 64x32 A tile + 64x32 B tile: 4 async b128 instructions per wave
    auto stage = [&](int buf, int kc) {
        for (int it = 0; it < 2; ++it) {
            int j   = tid + 128 * it;     // 0..255 transfers per tile
            int row = j >> 2;
            int seg = (j & 3) * 8;
            async_load_b128((unsigned)(uintptr_t)&lds_a[buf][row * 32 + seg],
                            A + (size_t)(row0 + row) * K + kc + seg);
            async_load_b128((unsigned)(uintptr_t)&lds_b[buf][row * 32 + seg],
                            W + (size_t)(col0 + row) * K + kc + seg);
        }
    };

    stage(0, 0);
    int buf = 0;
    for (int kc = 0; kc < K; kc += 32) {
        if (kc + 32 < K) {
            stage(buf ^ 1, kc + 32);                           // prefetch next tile
            asm volatile("s_wait_asynccnt 0x4" ::: "memory");  // current tile landed
        } else {
            asm volatile("s_wait_asynccnt 0x0" ::: "memory");
        }
        __syncthreads();
        // batch all fragment loads -> one DS clause / one dscnt wait,
        // then 4 back-to-back WMMAs
        const unsigned short* la = &lds_a[buf][(wave * 16 + r) * 32 + koff];
        v16bf af = mk16(*(const v8bf*)(la), *(const v8bf*)(la + 16));
        v16bf bfr[4];
        for (int t = 0; t < 4; ++t) {
            const unsigned short* lb = &lds_b[buf][(16 * t + r) * 32 + koff];
            bfr[t] = mk16(*(const v8bf*)(lb), *(const v8bf*)(lb + 16));
        }
        for (int t = 0; t < 4; ++t)
            acc[t] = __builtin_amdgcn_wmma_f32_16x16x32_bf16(
                false, af, false, bfr[t], (short)0, acc[t], false, false);
        __syncthreads();
        buf ^= 1;
    }

    for (int t = 0; t < 4; ++t) {
        int col = col0 + 16 * t + (lane & 15);
        float b = (vis && bias_vis) ? bias_vis[col] : 0.0f;
        for (int cr = 0; cr < 8; ++cr)
            C[(size_t)(row0 + wave * 16 + cr + mhalf) * N + col] = acc[t][cr] + b;
    }
}

// ---------------- RoPE + pack to WMMA layouts ----------------
// qkv[S,3*HID] f32 -> qp/kp [H,S,D] bf16 (q pre-scaled), vp [H,D,S] bf16
__global__ void rope_pack(const float* __restrict__ qkv,
                          const float* __restrict__ cosT, const float* __restrict__ sinT,
                          unsigned short* __restrict__ qp, unsigned short* __restrict__ kp,
                          unsigned short* __restrict__ vp) {
    int idx = blockIdx.x * blockDim.x + threadIdx.x;   // S*H*D threads
    int d = idx & (HDIM - 1);
    int h = (idx >> 7) & (HEADS - 1);
    int s = idx >> 12;
    const float* row = qkv + (size_t)s * (3 * HID);
    float c = cosT[s * HDIM + d], sn = sinT[s * HDIM + d];
    float q = row[h * HDIM + d];
    float k = row[HID + h * HDIM + d];
    float v = row[2 * HID + h * HDIM + d];
    float qr, kr;
    if (d < HDIM / 2) { qr = -row[h * HDIM + d + 64]; kr = -row[HID + h * HDIM + d + 64]; }
    else              { qr =  row[h * HDIM + d - 64]; kr =  row[HID + h * HDIM + d - 64]; }
    float qo = (q * c + qr * sn) * QKSCALE;
    float ko =  k * c + kr * sn;
    qp[((size_t)h * S_TOK + s) * HDIM + d] = f2bf(qo);
    kp[((size_t)h * S_TOK + s) * HDIM + d] = f2bf(ko);
    vp[((size_t)h * HDIM + d) * S_TOK + s] = f2bf(v);
}

// ---------------- flash attention: one wave per (16-query tile, head) ----------------
__global__ void flash_attn(const unsigned short* __restrict__ qp,
                           const unsigned short* __restrict__ kp,
                           const unsigned short* __restrict__ vp,
                           unsigned short* __restrict__ attn /* [S,HID] bf16 */) {
    __shared__ alignas(16) unsigned short plds[16 * 40];   // 16x32 P tile, padded stride
    const int lane  = threadIdx.x;
    const int q0    = blockIdx.x * 16;
    const int h     = blockIdx.y;
    const int r     = lane & 15;
    const int koff  = (lane >> 4) << 3;
    const int mhalf = (lane >> 4) << 3;

    // Q fragments (scale pre-folded)
    const unsigned short* qrow = qp + ((size_t)h * S_TOK + q0 + r) * HDIM;
    v16bf aq[4];
    for (int c = 0; c < 4; ++c)
        aq[c] = mk16(*(const v8bf*)(qrow + 32 * c + koff),
                     *(const v8bf*)(qrow + 32 * c + koff + 16));

    v8f o[8] = {};
    float mrow[8], lrow[8];
    for (int i = 0; i < 8; ++i) { mrow[i] = -1.0e30f; lrow[i] = 0.0f; }

    const unsigned short* kbase = kp + (size_t)h * S_TOK * HDIM;
    const unsigned short* vbase = vp + (size_t)h * HDIM * S_TOK;

    for (int key0 = 0; key0 <= q0 + 15; key0 += 32) {
        const unsigned short* k0row = kbase + (size_t)(key0 + r) * HDIM;
        const unsigned short* k1row = kbase + (size_t)(key0 + 16 + r) * HDIM;
        __builtin_prefetch(k0row + 32 * HDIM);     // next key block
        __builtin_prefetch(k1row + 32 * HDIM);
        // S = Q K^T for 32 keys (two 16-col tiles)
        v8f s0 = {}, s1 = {};
        for (int c = 0; c < 4; ++c) {
            v16bf b0 = mk16(*(const v8bf*)(k0row + 32 * c + koff),
                            *(const v8bf*)(k0row + 32 * c + koff + 16));
            v16bf b1 = mk16(*(const v8bf*)(k1row + 32 * c + koff),
                            *(const v8bf*)(k1row + 32 * c + koff + 16));
            s0 = __builtin_amdgcn_wmma_f32_16x16x32_bf16(false, aq[c], false, b0,
                                                         (short)0, s0, false, false);
            s1 = __builtin_amdgcn_wmma_f32_16x16x32_bf16(false, aq[c], false, b1,
                                                         (short)0, s1, false, false);
        }
        // causal mask (C layout: lane -> col, vgpr+half -> row)
        for (int cr = 0; cr < 8; ++cr) {
            int qidx = q0 + cr + mhalf;
            int kid  = key0 + (lane & 15);
            if (kid      > qidx) s0[cr] = -1.0e30f;
            if (kid + 16 > qidx) s1[cr] = -1.0e30f;
        }
        // online softmax: 16-lane row reductions
        float p0[8], p1[8];
        for (int cr = 0; cr < 8; ++cr) {
            float v = fmaxf(s0[cr], s1[cr]);
            for (int msk = 1; msk < 16; msk <<= 1) v = fmaxf(v, __shfl_xor(v, msk, 32));
            float mnew  = fmaxf(mrow[cr], v);
            float alpha = __expf(mrow[cr] - mnew);
            mrow[cr] = mnew;
            p0[cr] = __expf(s0[cr] - mnew);
            p1[cr] = __expf(s1[cr] - mnew);
            float sum = p0[cr] + p1[cr];
            for (int msk = 1; msk < 16; msk <<= 1) sum += __shfl_xor(sum, msk, 32);
            lrow[cr] = lrow[cr] * alpha + sum;
            for (int t = 0; t < 8; ++t) o[t][cr] *= alpha;
        }
        // C-layout -> A-layout via LDS
        for (int cr = 0; cr < 8; ++cr) {
            int row = cr + mhalf, col = lane & 15;
            plds[row * 40 + col]      = f2bf(p0[cr]);
            plds[row * 40 + col + 16] = f2bf(p1[cr]);
        }
        asm volatile("s_wait_dscnt 0" ::: "memory");
        v16bf ap = mk16(*(const v8bf*)(&plds[r * 40 + koff]),
                        *(const v8bf*)(&plds[r * 40 + koff + 16]));
        // O += P V  (V transposed [H,D,S]: keys contiguous);
        // load V fragments in groups of 4 so one loadcnt wait covers 4 WMMAs
        for (int g = 0; g < 2; ++g) {
            v16bf bv[4];
            for (int u = 0; u < 4; ++u) {
                const unsigned short* vcol =
                    vbase + (size_t)(16 * (4 * g + u) + r) * S_TOK + key0;
                bv[u] = mk16(*(const v8bf*)(vcol + koff),
                             *(const v8bf*)(vcol + koff + 16));
            }
            for (int u = 0; u < 4; ++u)
                o[4 * g + u] = __builtin_amdgcn_wmma_f32_16x16x32_bf16(
                    false, ap, false, bv[u], (short)0, o[4 * g + u], false, false);
        }
        asm volatile("s_wait_dscnt 0" ::: "memory");  // protect plds before next overwrite
    }
    // normalize + store bf16 [S,HID]
    for (int t = 0; t < 8; ++t) {
        int col = h * HDIM + 16 * t + (lane & 15);
        for (int cr = 0; cr < 8; ++cr) {
            float val = o[t][cr] / lrow[cr];
            attn[(size_t)(q0 + cr + mhalf) * HID + col] = f2bf(val);
        }
    }
}

// ---------------- host-side orchestration ----------------
static inline size_t alignup(size_t x) { return (x + 255) & ~(size_t)255; }

extern "C" void kernel_launch(void* const* d_in, const int* in_sizes, int n_in,
                              void* d_out, int out_size, void* d_ws, size_t ws_size,
                              hipStream_t stream) {
    const float* hidden  = (const float*)d_in[0];
    const float* cosT    = (const float*)d_in[1];
    const float* sinT    = (const float*)d_in[2];
    const float* wq_lang = (const float*)d_in[3];   // [HID, 3*HID]
    const float* wq_vis  = (const float*)d_in[4];
    const float* bq_vis  = (const float*)d_in[5];   // [3*HID]
    const float* wd_lang = (const float*)d_in[6];   // [HID, HID]
    const float* wd_vis  = (const float*)d_in[7];
    float* out = (float*)d_out;
    (void)in_sizes; (void)n_in; (void)out_size; (void)ws_size;

    char* ws = (char*)d_ws;
    size_t off = 0;
    unsigned short* hx    = (unsigned short*)(ws + off); off = alignup(off + (size_t)S_TOK * HID * 2);
    unsigned short* wqlT  = (unsigned short*)(ws + off); off = alignup(off + (size_t)3 * HID * HID * 2);
    unsigned short* wqvT  = (unsigned short*)(ws + off); off = alignup(off + (size_t)3 * HID * HID * 2);
    unsigned short* wdlT  = (unsigned short*)(ws + off); off = alignup(off + (size_t)HID * HID * 2);
    unsigned short* wdvT  = (unsigned short*)(ws + off); off = alignup(off + (size_t)HID * HID * 2);
    float*          qkv   = (float*)(ws + off);          off = alignup(off + (size_t)S_TOK * 3 * HID * 4);
    unsigned short* qp    = (unsigned short*)(ws + off); off = alignup(off + (size_t)HEADS * S_TOK * HDIM * 2);
    unsigned short* kp    = (unsigned short*)(ws + off); off = alignup(off + (size_t)HEADS * S_TOK * HDIM * 2);
    unsigned short* vp    = (unsigned short*)(ws + off); off = alignup(off + (size_t)HEADS * HDIM * S_TOK * 2);
    unsigned short* attnb = (unsigned short*)(ws + off); off = alignup(off + (size_t)S_TOK * HID * 2);

    // 1) precision prepass: hidden + weights to bf16 (weights transposed to [N,K])
    {
        int n = S_TOK * HID;
        cvt_bf16<<<(n + 255) / 256, 256, 0, stream>>>(hidden, hx, n);
        transpose_bf16<<<dim3(3 * HID / 32, HID / 32), 256, 0, stream>>>(wq_lang, wqlT, HID, 3 * HID);
        transpose_bf16<<<dim3(3 * HID / 32, HID / 32), 256, 0, stream>>>(wq_vis,  wqvT, HID, 3 * HID);
        transpose_bf16<<<dim3(HID / 32, HID / 32),     256, 0, stream>>>(wd_lang, wdlT, HID, HID);
        transpose_bf16<<<dim3(HID / 32, HID / 32),     256, 0, stream>>>(wd_vis,  wdvT, HID, HID);
    }
    // 2) modality-routed QKV GEMM (vision bias in epilogue)
    gemm_routed<<<dim3(S_TOK / 64, 3 * HID / 64), 128, 0, stream>>>(
        hx, wqlT, wqvT, bq_vis, qkv, 3 * HID, HID);
    // 3) RoPE + pack (q pre-scaled by 1/sqrt(D); v transposed)
    rope_pack<<<(S_TOK * HEADS * HDIM) / 256, 256, 0, stream>>>(qkv, cosT, sinT, qp, kp, vp);
    // 4) causal flash attention
    flash_attn<<<dim3(S_TOK / 16, HEADS), 32, 0, stream>>>(qp, kp, vp, attnb);
    // 5) modality-routed dense projection -> fp32 output
    gemm_routed<<<dim3(S_TOK / 64, HID / 64), 128, 0, stream>>>(
        attnb, wdlT, wdvT, nullptr, out, HID, HID);
}
